// SRNN_Layer_5549097746580
// MI455X (gfx1250) — matrix operand order repaired and checked by reading, not compiled, not run on hardware
//
#include <hip/hip_runtime.h>
#include <hip/hip_bf16.h>
#include <math.h>

// Problem sizes (fixed by the reference).
#define B_SZ   64
#define T_SZ   512
#define DIN    512
#define DH     1024

typedef __bf16 v16bf __attribute__((ext_vector_type(16)));
typedef float  v8f   __attribute__((ext_vector_type(8)));

// ---------------------------------------------------------------------------
// A-operand loader: 16x32 bf16 tile slice for one lane.
// ISA layout (16-bit A 16x32): lane l -> row m = l&15, khalf = l>>4.
//   v16bf element e (0..7)  -> k = khalf*8 + e
//   v16bf element e (8..15) -> k = 16 + khalf*8 + (e-8)
// Source rows are contiguous f32, so this is 4x float4 loads + cvt.
// ---------------------------------------------------------------------------
__device__ __forceinline__ v16bf load_a_tile(const float* __restrict__ A, int lda,
                                             int row_base, int kc, int lane) {
  const int m     = row_base + (lane & 15);
  const int khalf = lane >> 4;
  const float* p  = A + (size_t)m * lda + kc * 32 + khalf * 8;
  const float4 a0 = *reinterpret_cast<const float4*>(p);
  const float4 a1 = *reinterpret_cast<const float4*>(p + 4);
  const float4 a2 = *reinterpret_cast<const float4*>(p + 16);
  const float4 a3 = *reinterpret_cast<const float4*>(p + 20);
  v16bf r;
  r[0]  = (__bf16)a0.x; r[1]  = (__bf16)a0.y; r[2]  = (__bf16)a0.z; r[3]  = (__bf16)a0.w;
  r[4]  = (__bf16)a1.x; r[5]  = (__bf16)a1.y; r[6]  = (__bf16)a1.z; r[7]  = (__bf16)a1.w;
  r[8]  = (__bf16)a2.x; r[9]  = (__bf16)a2.y; r[10] = (__bf16)a2.z; r[11] = (__bf16)a2.w;
  r[12] = (__bf16)a3.x; r[13] = (__bf16)a3.y; r[14] = (__bf16)a3.z; r[15] = (__bf16)a3.w;
  return r;
}

// ---------------------------------------------------------------------------
// Pack a KxN f32 matrix into bf16 B-operand tiles, pre-swizzled per lane:
// B 32x16 bf16 layout: lane l -> col n = l&15, khalf = l>>4; element e -> k =
// khalf*8 + e (e<8), 16 + khalf*8 + (e-8) (e>=8). Packed index = ((ntile*kchunks + kc)*32 + lane), 16 elems each,
// so the GEMM loops read one contiguous, coalesced 32B vector per lane.
// ---------------------------------------------------------------------------
__global__ void pack_b_bf16_kernel(const float* __restrict__ src,
                                   __bf16* __restrict__ dst, int K, int N) {
  const int kchunks = K >> 5;            // K / 32
  const int total   = (N >> 4) * kchunks * 32;
  const int tid = blockIdx.x * blockDim.x + threadIdx.x;
  if (tid >= total) return;
  const int lane  = tid & 31;
  const int kc    = (tid >> 5) % kchunks;
  const int ntile = (tid >> 5) / kchunks;
  const int n  = ntile * 16 + (lane & 15);
  const int k0 = kc * 32 + (lane >> 4) * 8;
  v16bf v;
#pragma unroll
  for (int e = 0; e < 16; ++e) v[e] = (__bf16)src[(size_t)(k0 + (e < 8 ? e : e + 8)) * N + n];
  reinterpret_cast<v16bf*>(dst)[tid] = v;
}

// ---------------------------------------------------------------------------
// Input projection: out[r, :] = x[r, :] @ W + bias, r in [0, B*T).
// One wave computes one 16-row M tile x 4 N tiles (4 accumulators), K = 512.
// ---------------------------------------------------------------------------
#define XW_NT 4
__global__ void xw_gemm_kernel(const float* __restrict__ x,
                               const float* __restrict__ bias,
                               const __bf16* __restrict__ Wp,
                               float* __restrict__ out) {
  const int wave = (blockIdx.x * blockDim.x + threadIdx.x) >> 5;
  const int lane = threadIdx.x & 31;
  const int ngroups = (DH / 16) / XW_NT;       // 16
  const int mtile = wave / ngroups;            // 0..2047
  const int ng    = wave % ngroups;
  const int kchunks = DIN / 32;                // 16

  v8f acc[XW_NT] = {};
  const v16bf* Bv = reinterpret_cast<const v16bf*>(Wp);

  for (int kc = 0; kc < kchunks; ++kc) {
    const v16bf a = load_a_tile(x, DIN, mtile * 16, kc, lane);
#pragma unroll
    for (int j = 0; j < XW_NT; ++j) {
      const int ntile = ng * XW_NT + j;
      const v16bf b = Bv[(size_t)(ntile * kchunks + kc) * 32 + lane];
      acc[j] = __builtin_amdgcn_wmma_f32_16x16x32_bf16(
          false, a, false, b, (short)0, acc[j], false, false);
    }
  }

  // C/D layout: v8f element r -> row m = (lane>>4)*8 + r, col n = lane&15.
  const int mb = (lane >> 4) * 8;
  const int n0 = lane & 15;
  const int row_base = mtile * 16;
#pragma unroll
  for (int j = 0; j < XW_NT; ++j) {
    const int col = (ng * XW_NT + j) * 16 + n0;
    const float bv = bias[col];
#pragma unroll
    for (int r = 0; r < 8; ++r) {
      out[(size_t)(row_base + mb + r) * DH + col] = acc[j][r] + bv;
    }
  }
}

// ---------------------------------------------------------------------------
// Persistent recurrence kernel: all 512 steps in one launch.
//   Grid: NBLK=32 blocks x 128 threads (4 waves). Block b owns output columns
//   [b*32, b*32+32) (2 N tiles); wave w owns M tile w (rows w*16..w*16+15).
//   The block's slice of packed-bf16 R (64 KB) is staged into LDS once and
//   reused by all 512 steps (ds_load instead of 512 x 2MB from L2).
//   Steps are separated by a grid-wide monotonic-counter barrier.
//   slice t of d_out is read (pre-activation), updated to tanh(pre + h@R)
//   in place, and becomes h for step t+1.
// ---------------------------------------------------------------------------
#define NBLK         32
#define SCAN_THREADS 128

__global__ void __launch_bounds__(SCAN_THREADS)
rnn_scan_kernel(const float* __restrict__ h0,
                float* __restrict__ out,
                const __bf16* __restrict__ Rp,
                unsigned* __restrict__ bar) {
  __shared__ __bf16 Bs[2 * (DH / 32) * 32 * 16];   // 64 KB packed R columns
  const int tid  = threadIdx.x;
  const int blk  = blockIdx.x;       // 0..31
  const int wave = tid >> 5;         // 0..3 -> M tile
  const int lane = tid & 31;
  const int kchunks = DH / 32;       // 32

  // One-time stage: this block's 2 N tiles of packed R -> LDS (coalesced).
  {
    const v16bf* src = reinterpret_cast<const v16bf*>(Rp)
                     + (size_t)(blk * 2) * kchunks * 32;
    v16bf* dst = reinterpret_cast<v16bf*>(Bs);
    for (int i = tid; i < 2 * kchunks * 32; i += SCAN_THREADS) dst[i] = src[i];
  }
  __syncthreads();

  const v16bf* Bl = reinterpret_cast<const v16bf*>(Bs);
  const int mtile = wave;
  const int mb = (lane >> 4) * 8;
  const int n0 = lane & 15;

  for (int t = 0; t < T_SZ; ++t) {
    const float* hsrc = (t == 0) ? h0 : out + (size_t)(t - 1) * DH;
    const int ld      = (t == 0) ? DH : T_SZ * DH;

    v8f acc[2] = {};
    for (int kc = 0; kc < kchunks; ++kc) {
      const v16bf a = load_a_tile(hsrc, ld, mtile * 16, kc, lane);
#pragma unroll
      for (int j = 0; j < 2; ++j) {
        const v16bf b = Bl[(size_t)(j * kchunks + kc) * 32 + lane];
        acc[j] = __builtin_amdgcn_wmma_f32_16x16x32_bf16(
            false, a, false, b, (short)0, acc[j], false, false);
      }
    }

    float* slice = out + (size_t)t * DH;
#pragma unroll
    for (int j = 0; j < 2; ++j) {
      const int col = (blk * 2 + j) * 16 + n0;
#pragma unroll
      for (int r = 0; r < 8; ++r) {
        const size_t idx = (size_t)(mtile * 16 + mb + r) * ((size_t)T_SZ * DH) + col;
        slice[idx] = tanhf(slice[idx] + acc[j][r]);
      }
    }

    // Grid-wide barrier: next step reads this slice as h from other blocks.
    if (t + 1 < T_SZ) {
      __threadfence();               // release our slice stores (device scope)
      __syncthreads();               // all waves in block fenced
      if (tid == 0) {
        __hip_atomic_fetch_add(bar, 1u, __ATOMIC_RELEASE,
                               __HIP_MEMORY_SCOPE_AGENT);
        const unsigned target = (unsigned)(t + 1) * NBLK;
        while (__hip_atomic_load(bar, __ATOMIC_ACQUIRE,
                                 __HIP_MEMORY_SCOPE_AGENT) < target)
          __builtin_amdgcn_s_sleep(1);
      }
      __syncthreads();               // acquire-inv by wave 0 covers the WGP$
    }
  }
}

__global__ void zero_u32_kernel(unsigned* p) {
  if (threadIdx.x == 0 && blockIdx.x == 0) *p = 0u;
}

// ---------------------------------------------------------------------------
// Launch: pack W,R -> xW GEMM (pre-activations into d_out) -> zero barrier ->
// one persistent scan kernel covering all 512 steps.
// Deterministic: barrier counter re-zeroed every call; d_out fully rewritten.
// ---------------------------------------------------------------------------
extern "C" void kernel_launch(void* const* d_in, const int* in_sizes, int n_in,
                              void* d_out, int out_size, void* d_ws, size_t ws_size,
                              hipStream_t stream) {
  const float* x    = (const float*)d_in[0];
  const float* h0   = (const float*)d_in[1];
  const float* W    = (const float*)d_in[2];
  const float* R    = (const float*)d_in[3];
  const float* bias = (const float*)d_in[4];
  float* out = (float*)d_out;

  __bf16*   Wp  = (__bf16*)d_ws;                                   // 1 MB
  __bf16*   Rp  = (__bf16*)((char*)d_ws + (size_t)DIN * DH * 2);   // 2 MB
  unsigned* bar = (unsigned*)((char*)d_ws + (size_t)(DIN + DH) * DH * 2);

  // Pack W: (DH/16)*(DIN/32)*32 = 32768 lane-vectors.
  pack_b_bf16_kernel<<<32768 / 256, 256, 0, stream>>>(W, Wp, DIN, DH);
  // Pack R: (DH/16)*(DH/32)*32 = 65536 lane-vectors.
  pack_b_bf16_kernel<<<65536 / 256, 256, 0, stream>>>(R, Rp, DH, DH);

  // xW + b -> d_out. waves = (B*T/16) * ((DH/16)/XW_NT) = 2048*16 = 32768
  //  -> 1,048,576 threads -> 4096 blocks of 256.
  xw_gemm_kernel<<<4096, 256, 0, stream>>>(x, bias, Wp, out);

  // Reset grid barrier, then run the whole scan in one persistent kernel.
  zero_u32_kernel<<<1, 64, 0, stream>>>(bar);
  rnn_scan_kernel<<<NBLK, SCAN_THREADS, 0, stream>>>(h0, out, Rp, bar);
}
